// EntityLinker_76605036691535
// MI455X (gfx1250) — compile-verified, matrix operand order checked by reading, not run on hardware
//
#include <hip/hip_runtime.h>
#include <stdint.h>

typedef __attribute__((ext_vector_type(16))) __bf16 bf16x16;
typedef __attribute__((ext_vector_type(8)))  __bf16 bf16x8;
typedef __attribute__((ext_vector_type(8)))  float  f32x8;

namespace {
constexpr int cB = 4, cL = 2048, cD = 512, cH = 8, cHD = 64;
constexpr int cNL = 2, cDFF = 2048, cK = 5, cNC = 64, cMAXM = 10;
constexpr int cBL = cB * cL;  // 8192 rows
}

__device__ __forceinline__ f32x8 wmma_bf16(bf16x16 a, bf16x16 b, f32x8 c) {
  return __builtin_amdgcn_wmma_f32_16x16x32_bf16(false, a, false, b, (short)0, c,
                                                 false, false);
}

// CDNA5 async memory->LDS DMA (ASYNCcnt-tracked); one 16B chunk per lane.
__device__ __forceinline__ void async_ld16(uint32_t lds_addr, const __bf16* gptr) {
  asm volatile("global_load_async_to_lds_b128 %0, %1, off"
               :: "v"(lds_addr), "v"(gptr) : "memory");
}
__device__ __forceinline__ void wait_async0() {
  asm volatile("s_wait_asynccnt 0x0" ::: "memory");
}
// In-order async completion: residue of 6 == previous slab (6 chunks/thread) done.
__device__ __forceinline__ void wait_async6() {
  asm volatile("s_wait_asynccnt 0x6" ::: "memory");
}

// ------------------------------------------------------------------
// Generic NT GEMM: C[MxN] = act(alpha * A[MxK] @ B[NxK]^T + bias[N])
// A,B bf16 row-major (K contiguous). Optional f32 and/or bf16 output.
// Block tile 128x64, K slab 64; 8 waves, each wave 32x32 via 2x2 WMMA
// fragments, 8 WMMAs per barrier round. Double-buffered LDS staging:
// full tiles prefetch slab k+1 via GLOBAL_LOAD_ASYNC_TO_LDS_B128 while
// slab k is consumed (latency hidden behind WMMAs). Ragged tiles fall
// back to masked load + ds_store (zero-filled out-of-range rows).
// ------------------------------------------------------------------
#define GBM 128
#define GBN 64
#define GBK 64
#define GLS 72   // LDS row stride in halves (64 + 8 pad) -> 144B, 16B aligned
#define GSLAB ((GBM + GBN) * GLS)  // one slab in halves (A tile + B tile)

__device__ __forceinline__ void stage_slab(
    const __bf16* __restrict__ A, int lda, const __bf16* __restrict__ B, int ldb,
    int M, int N, int m0, int n0, int k0, bool fullA, bool fullB, int tid,
    __bf16* As, __bf16* Bs, uint32_t ldsA0, uint32_t ldsB0) {
  // A tile 128x64 halves (16KB): 1024 16B chunks, 4 per thread.
  if (fullA) {
#pragma unroll
    for (int i = 0; i < 4; ++i) {
      int c   = tid + 256 * i;
      int row = c >> 3;
      int cc  = (c & 7) * 8;
      async_ld16(ldsA0 + (uint32_t)(row * GLS + cc) * 2,
                 A + (size_t)(m0 + row) * lda + k0 + cc);
    }
  } else {
#pragma unroll
    for (int i = 0; i < 4; ++i) {
      int c   = tid + 256 * i;
      int row = c >> 3;
      int cc  = (c & 7) * 8;
      bf16x8 v;
#pragma unroll
      for (int e = 0; e < 8; ++e) v[e] = (__bf16)0.0f;
      if (m0 + row < M) v = *(const bf16x8*)(A + (size_t)(m0 + row) * lda + k0 + cc);
      *(bf16x8*)(&As[row * GLS + cc]) = v;
    }
  }
  // B tile 64x64 halves (8KB): 512 16B chunks, 2 per thread.
  if (fullB) {
#pragma unroll
    for (int i = 0; i < 2; ++i) {
      int c   = tid + 256 * i;
      int row = c >> 3;
      int cc  = (c & 7) * 8;
      async_ld16(ldsB0 + (uint32_t)(row * GLS + cc) * 2,
                 B + (size_t)(n0 + row) * ldb + k0 + cc);
    }
  } else {
#pragma unroll
    for (int i = 0; i < 2; ++i) {
      int c   = tid + 256 * i;
      int row = c >> 3;
      int cc  = (c & 7) * 8;
      bf16x8 v;
#pragma unroll
      for (int e = 0; e < 8; ++e) v[e] = (__bf16)0.0f;
      if (n0 + row < N) v = *(const bf16x8*)(B + (size_t)(n0 + row) * ldb + k0 + cc);
      *(bf16x8*)(&Bs[row * GLS + cc]) = v;
    }
  }
}

__global__ __launch_bounds__(256) void gemm_bf16_nt(
    const __bf16* __restrict__ A, int lda, const __bf16* __restrict__ B, int ldb,
    const float* __restrict__ bias, float alpha, int do_relu,
    float* __restrict__ Cf, int ldcf, __bf16* __restrict__ Ch, int ldch,
    int M, int N, int Kd) {
  __shared__ char lds_raw[2 * GSLAB * 2];  // two slabs, ~55KB
  const uint32_t lds0 = (uint32_t)(uintptr_t)lds_raw;  // LDS aperture: low 32b = offset

  const int tid  = threadIdx.x;
  const int m0   = blockIdx.x * GBM;
  const int n0   = blockIdx.y * GBN;
  const int lane = tid & 31;
  const int wave = tid >> 5;
  const int wm   = (wave & 3) * 32;   // 4 waves along M
  const int wn   = (wave >> 2) * 32;  // 2 waves along N
  const int hlf  = lane >> 4;         // lane half selects K sub-slice
  const int mr   = lane & 15;         // row (A) / col (B/C) within fragment
  const int koff = hlf * 8;
  const bool fullA = (m0 + GBM <= M);
  const bool fullB = (n0 + GBN <= N);
  const bool allasync = fullA && fullB;

  f32x8 acc[2][2];
#pragma unroll
  for (int i = 0; i < 2; ++i)
#pragma unroll
    for (int j = 0; j < 2; ++j)
#pragma unroll
      for (int e = 0; e < 8; ++e) acc[i][j][e] = 0.0f;

  // prologue: stage slab 0 into buffer 0
  stage_slab(A, lda, B, ldb, M, N, m0, n0, 0, fullA, fullB, tid,
             (__bf16*)lds_raw, (__bf16*)lds_raw + GBM * GLS,
             lds0, lds0 + (uint32_t)GBM * GLS * 2);

  int buf = 0;
  for (int k0 = 0; k0 < Kd; k0 += GBK) {
    const bool haveNext = (k0 + GBK < Kd);
    if (haveNext) {
      int nb = buf ^ 1;
      stage_slab(A, lda, B, ldb, M, N, m0, n0, k0 + GBK, fullA, fullB, tid,
                 (__bf16*)lds_raw + (size_t)nb * GSLAB,
                 (__bf16*)lds_raw + (size_t)nb * GSLAB + GBM * GLS,
                 lds0 + (uint32_t)(nb * GSLAB * 2),
                 lds0 + (uint32_t)(nb * GSLAB * 2 + GBM * GLS * 2));
    }
    if (allasync) {
      if (haveNext) wait_async6();  // current slab done; next slab still in flight
      else          wait_async0();
    } else if (fullA || fullB) {
      wait_async0();
    }
    __syncthreads();

    __bf16* As = (__bf16*)lds_raw + (size_t)buf * GSLAB;
    __bf16* Bs = As + GBM * GLS;
    // ---- compute: two K=32 sub-slabs, 4 WMMAs each ----
#pragma unroll
    for (int ks = 0; ks < GBK; ks += 32) {
      bf16x16 af[2], bfr[2];
#pragma unroll
      for (int i = 0; i < 2; ++i) {
        int row = wm + i * 16 + mr;
        bf16x8 lo = *(const bf16x8*)(&As[row * GLS + ks + koff]);
        bf16x8 hi = *(const bf16x8*)(&As[row * GLS + ks + 16 + koff]);
        bf16x16 t;
#pragma unroll
        for (int e = 0; e < 8; ++e) { t[e] = lo[e]; t[e + 8] = hi[e]; }
        af[i] = t;
      }
#pragma unroll
      for (int j = 0; j < 2; ++j) {
        int row = wn + j * 16 + mr;
        bf16x8 lo = *(const bf16x8*)(&Bs[row * GLS + ks + koff]);
        bf16x8 hi = *(const bf16x8*)(&Bs[row * GLS + ks + 16 + koff]);
        bf16x16 t;
#pragma unroll
        for (int e = 0; e < 8; ++e) { t[e] = lo[e]; t[e + 8] = hi[e]; }
        bfr[j] = t;
      }
#pragma unroll
      for (int i = 0; i < 2; ++i)
#pragma unroll
        for (int j = 0; j < 2; ++j) acc[i][j] = wmma_bf16(af[i], bfr[j], acc[i][j]);
    }
    __syncthreads();
    buf ^= 1;
  }

  // epilogue: C layout -> VGPR v holds rows {v, v+8} (lane halves), col = lane&15
#pragma unroll
  for (int i = 0; i < 2; ++i)
#pragma unroll
    for (int j = 0; j < 2; ++j) {
      int n    = n0 + wn + j * 16 + mr;
      float bv = (bias && n < N) ? bias[n] : 0.0f;
#pragma unroll
      for (int vv = 0; vv < 8; ++vv) {
        int m = m0 + wm + i * 16 + vv + 8 * hlf;
        if (m < M && n < N) {
          float val = alpha * acc[i][j][vv] + bv;
          if (do_relu) val = fmaxf(val, 0.0f);
          if (Cf) Cf[(size_t)m * ldcf + n] = val;
          if (Ch) Ch[(size_t)m * ldch + n] = (__bf16)val;
        }
      }
    }
}

// ------------------------------------------------------------------
__global__ __launch_bounds__(256) void cvt_bf16_kernel(const float* __restrict__ s,
                                                       __bf16* __restrict__ d, int n) {
  int i = blockIdx.x * 256 + threadIdx.x;
  if (i < n) d[i] = (__bf16)s[i];
}

__global__ __launch_bounds__(256) void embed_kernel(
    const int* __restrict__ ids, const float* __restrict__ tok,
    const float* __restrict__ pos, float* __restrict__ xf, __bf16* __restrict__ xb) {
  int idx = blockIdx.x * 256 + threadIdx.x;  // exact grid: B*L*D
  int d  = idx & (cD - 1);
  int bl = idx >> 9;
  int l  = bl & (cL - 1);
  float v = tok[(size_t)ids[bl] * cD + d] + pos[(size_t)l * cD + d];
  xf[idx] = v;
  xb[idx] = (__bf16)v;
}

__global__ __launch_bounds__(256) void vt_kernel(const __bf16* __restrict__ qkv,
                                                 __bf16* __restrict__ vt) {
  int idx = blockIdx.x * 256 + threadIdx.x;  // exact grid: B*H*HD*L
  int l = idx & (cL - 1);
  int t = idx >> 11;
  int d = t & (cHD - 1); t >>= 6;
  int h = t & (cH - 1);
  int b = t >> 3;
  vt[idx] = qkv[((size_t)(b * cL + l)) * (3 * cD) + 2 * cD + h * cHD + d];
}

__global__ __launch_bounds__(256) void softmax_kernel(const float* __restrict__ S,
                                                      __bf16* __restrict__ P, int n) {
  __shared__ float red[8];
  const int row = blockIdx.x;
  const float* sr = S + (size_t)row * n;
  __bf16* pr = P + (size_t)row * n;
  int lane = threadIdx.x & 31, wave = threadIdx.x >> 5;
  float mx = -3.0e38f;
  for (int c = threadIdx.x; c < n; c += 256) mx = fmaxf(mx, sr[c]);
  for (int m = 16; m >= 1; m >>= 1) mx = fmaxf(mx, __shfl_xor(mx, m, 32));
  if (lane == 0) red[wave] = mx;
  __syncthreads();
  mx = red[0];
  for (int i = 1; i < 8; ++i) mx = fmaxf(mx, red[i]);
  __syncthreads();
  float sum = 0.f;
  for (int c = threadIdx.x; c < n; c += 256) sum += __expf(sr[c] - mx);
  for (int m = 16; m >= 1; m >>= 1) sum += __shfl_xor(sum, m, 32);
  if (lane == 0) red[wave] = sum;
  __syncthreads();
  sum = 0.f;
  for (int i = 0; i < 8; ++i) sum += red[i];
  float inv = 1.0f / sum;
  for (int c = threadIdx.x; c < n; c += 256) pr[c] = (__bf16)(__expf(sr[c] - mx) * inv);
}

// LayerNorm(x + r) -> outF (f32, may alias x) and outH (bf16). Wave per row.
__global__ __launch_bounds__(256) void ln_kernel(
    const float* __restrict__ x, const float* __restrict__ r,
    const float* __restrict__ g, const float* __restrict__ beta,
    float* __restrict__ outF, __bf16* __restrict__ outH) {
  int lane = threadIdx.x & 31, wave = threadIdx.x >> 5;
  int row = blockIdx.x * 8 + wave;
  const float* xr = x + (size_t)row * cD;
  const float* rr = r + (size_t)row * cD;
  float v[16];
  float s = 0.f;
#pragma unroll
  for (int i = 0; i < 16; ++i) { v[i] = xr[lane + 32 * i] + rr[lane + 32 * i]; s += v[i]; }
  for (int m = 16; m >= 1; m >>= 1) s += __shfl_xor(s, m, 32);
  float mean = s * (1.0f / cD);
  float vs = 0.f;
#pragma unroll
  for (int i = 0; i < 16; ++i) { float d2 = v[i] - mean; vs += d2 * d2; }
  for (int m = 16; m >= 1; m >>= 1) vs += __shfl_xor(vs, m, 32);
  float rinv = rsqrtf(vs * (1.0f / cD) + 1e-5f);
#pragma unroll
  for (int i = 0; i < 16; ++i) {
    int d = lane + 32 * i;
    float o = (v[i] - mean) * rinv * g[d] + beta[d];
    outF[(size_t)row * cD + d] = o;
    outH[(size_t)row * cD + d] = (__bf16)o;
  }
}

__global__ __launch_bounds__(256) void logits_kernel(
    const float* __restrict__ hid, const float* __restrict__ sw,
    const float* __restrict__ sb, const float* __restrict__ ew,
    const float* __restrict__ eb, float* __restrict__ outS, float* __restrict__ outE) {
  int lane = threadIdx.x & 31, wave = threadIdx.x >> 5;
  int row = blockIdx.x * 8 + wave;
  const float* hr = hid + (size_t)row * cD;
  float a = 0.f, b2 = 0.f;
#pragma unroll
  for (int i = 0; i < 16; ++i) {
    int d = lane + 32 * i;
    float h = hr[d];
    a += h * sw[d];
    b2 += h * ew[d];
  }
  for (int m = 16; m >= 1; m >>= 1) { a += __shfl_xor(a, m, 32); b2 += __shfl_xor(b2, m, 32); }
  if (lane == 0) { outS[row] = a + sb[0]; outE[row] = b2 + eb[0]; }
}

// Top-5 per row (stable: ties -> lowest index). 8 waves: 4 start rows, 4 end rows.
__global__ __launch_bounds__(256) void topk_kernel(
    const float* __restrict__ sl, const float* __restrict__ el,
    int* __restrict__ sidx, int* __restrict__ eidx) {
  int lane = threadIdx.x & 31, wave = threadIdx.x >> 5;
  int b = wave & 3;
  const float* ptr = (wave < 4) ? (sl + (size_t)b * cL) : (el + (size_t)b * cL);
  int* outp = (wave < 4) ? (sidx + b * cK) : (eidx + b * cK);
  int chosen[cK];
  for (int k = 0; k < cK; ++k) {
    float best = -3.0e38f;
    int bidx = 0x7fffffff;
    for (int c = lane; c < cL; c += 32) {
      bool skip = false;
      for (int j = 0; j < k; ++j)
        if (chosen[j] == c) skip = true;
      if (skip) continue;
      float val = ptr[c];
      if (val > best || (val == best && c < bidx)) { best = val; bidx = c; }
    }
    for (int m = 16; m >= 1; m >>= 1) {
      float ov = __shfl_xor(best, m, 32);
      int oi = __shfl_xor(bidx, m, 32);
      if (ov > best || (ov == best && oi < bidx)) { best = ov; bidx = oi; }
    }
    chosen[k] = bidx;
    if (lane == 0) outp[k] = bidx;
  }
}

__global__ __launch_bounds__(256) void feat_kernel(
    const float* __restrict__ hid, const int* __restrict__ sidx,
    const int* __restrict__ eidx, __bf16* __restrict__ featb, int* __restrict__ valid) {
  int i = blockIdx.x;  // 0..B*K-1
  int b = i / cK;
  int s = sidx[i], e = eidx[i];
  if (threadIdx.x == 0) valid[i] = (e >= s && (e - s) < cMAXM) ? 1 : 0;
  const float* hs = hid + ((size_t)b * cL + s) * cD;
  const float* he = hid + ((size_t)b * cL + e) * cD;
  for (int d = threadIdx.x; d < cD; d += 256) {
    float a = hs[d], c = he[d];
    featb[(size_t)i * 3 * cD + d]          = (__bf16)a;
    featb[(size_t)i * 3 * cD + cD + d]     = (__bf16)c;
    featb[(size_t)i * 3 * cD + 2 * cD + d] = (__bf16)(a * c);
  }
}

__global__ __launch_bounds__(256) void compact_kernel(
    const float* __restrict__ raw, const int* __restrict__ valid,
    float* __restrict__ span_out, __bf16* __restrict__ spanb) {
  __shared__ int order[cK];
  __shared__ int keep[cK];
  int b = blockIdx.x;
  if (threadIdx.x == 0) {
    int p = 0;
    for (int k = 0; k < cK; ++k)
      if (valid[b * cK + k]) { order[p] = k; keep[p] = 1; ++p; }
    for (int k = 0; k < cK; ++k)
      if (!valid[b * cK + k]) { order[p] = k; keep[p] = 0; ++p; }
  }
  __syncthreads();
  for (int j = 0; j < cK; ++j) {
    int src = order[j];
    int kp = keep[j];
    for (int d = threadIdx.x; d < cD; d += 256) {
      float v = kp ? raw[((size_t)b * cK + src) * cD + d] : 0.0f;
      span_out[((size_t)b * cK + j) * cD + d] = v;
      spanb[((size_t)b * cK + j) * cD + d] = (__bf16)v;
    }
  }
}

// scores[b,m,c] = dot(mp[b,m,:], cp[b,m,c,:]); one wave per (b,m,c).
__global__ __launch_bounds__(256) void scores_kernel(
    const float* __restrict__ mp, const float* __restrict__ cp,
    float* __restrict__ out) {
  int lane = threadIdx.x & 31;
  int gw = (blockIdx.x * 256 + threadIdx.x) >> 5;  // 0..B*K*NC-1 (exact grid)
  int bm = gw >> 6;
  float s = 0.f;
#pragma unroll
  for (int i = 0; i < 16; ++i) {
    int d = lane + 32 * i;
    s += mp[(size_t)bm * cD + d] * cp[(size_t)gw * cD + d];
  }
  for (int m = 16; m >= 1; m >>= 1) s += __shfl_xor(s, m, 32);
  if (lane == 0) out[gw] = s;
}

// ------------------------------------------------------------------
extern "C" void kernel_launch(void* const* d_in, const int* in_sizes, int n_in,
                              void* d_out, int out_size, void* d_ws, size_t ws_size,
                              hipStream_t stream) {
  (void)in_sizes; (void)n_in; (void)out_size; (void)ws_size;
  const int* token_ids    = (const int*)d_in[0];
  const float* cand       = (const float*)d_in[1];
  const float* tok_emb    = (const float*)d_in[2];
  const float* pos_emb    = (const float*)d_in[3];
  const float* attn_in_w  = (const float*)d_in[4];
  const float* attn_in_b  = (const float*)d_in[5];
  const float* attn_out_w = (const float*)d_in[6];
  const float* attn_out_b = (const float*)d_in[7];
  const float* ln1_g = (const float*)d_in[8];
  const float* ln1_b = (const float*)d_in[9];
  const float* ffn_w1 = (const float*)d_in[10];
  const float* ffn_b1 = (const float*)d_in[11];
  const float* ffn_w2 = (const float*)d_in[12];
  const float* ffn_b2 = (const float*)d_in[13];
  const float* ln2_g = (const float*)d_in[14];
  const float* ln2_b = (const float*)d_in[15];
  const float* start_w = (const float*)d_in[16];
  const float* start_b = (const float*)d_in[17];
  const float* end_w = (const float*)d_in[18];
  const float* end_b = (const float*)d_in[19];
  const float* span_w1 = (const float*)d_in[20];
  const float* span_b1 = (const float*)d_in[21];
  const float* span_w2 = (const float*)d_in[22];
  const float* span_b2 = (const float*)d_in[23];
  const float* mention_w = (const float*)d_in[24];
  const float* mention_b = (const float*)d_in[25];
  const float* entity_w = (const float*)d_in[26];
  const float* entity_b = (const float*)d_in[27];

  float* out = (float*)d_out;
  const size_t SPAN_OFF = 0;      // B*K*D  = 10240
  const size_t SL_OFF = 10240;    // B*L    = 8192
  const size_t EL_OFF = 18432;    // B*L
  const size_t HID_OFF = 26624;   // B*L*D  = 4194304
  const size_t SC_OFF = 4220928;  // B*K*NC = 1280

  // workspace bump allocator (256B aligned)
  char* base = (char*)d_ws;
  size_t off = 0;
  auto alloc = [&](size_t bytes) -> void* {
    void* p = base + off;
    off = (off + bytes + 255) & ~(size_t)255;
    return p;
  };
  float*  xf    = (float*)alloc((size_t)cBL * cD * 4);
  __bf16* xb    = (__bf16*)alloc((size_t)cBL * cD * 2);
  float*  tmpf  = (float*)alloc((size_t)cBL * cD * 4);
  __bf16* qkvb  = (__bf16*)alloc((size_t)cBL * 3 * cD * 2);
  __bf16* vtb   = (__bf16*)alloc((size_t)cB * cH * cHD * cL * 2);
  __bf16* ob    = (__bf16*)alloc((size_t)cBL * cD * 2);
  __bf16* h1b   = (__bf16*)alloc((size_t)cBL * cDFF * 2);
  float*  sbuf  = (float*)alloc((size_t)cL * cL * 4);
  __bf16* pbuf  = (__bf16*)alloc((size_t)cL * cL * 2);
  __bf16* w_ain = (__bf16*)alloc((size_t)cNL * 3 * cD * cD * 2);
  __bf16* w_aout= (__bf16*)alloc((size_t)cNL * cD * cD * 2);
  __bf16* w_f1  = (__bf16*)alloc((size_t)cNL * cDFF * cD * 2);
  __bf16* w_f2  = (__bf16*)alloc((size_t)cNL * cD * cDFF * 2);
  __bf16* w_s1  = (__bf16*)alloc((size_t)cD * 3 * cD * 2);
  __bf16* w_s2  = (__bf16*)alloc((size_t)cD * cD * 2);
  __bf16* w_m   = (__bf16*)alloc((size_t)cD * cD * 2);
  __bf16* w_e   = (__bf16*)alloc((size_t)cD * cD * 2);
  __bf16* candb = (__bf16*)alloc((size_t)cB * cK * cNC * cD * 2);
  __bf16* featb = (__bf16*)alloc((size_t)cB * cK * 3 * cD * 2);
  __bf16* s1b   = (__bf16*)alloc((size_t)cB * cK * cD * 2);
  float*  spanraw = (float*)alloc((size_t)cB * cK * cD * 4);
  __bf16* spanb = (__bf16*)alloc((size_t)cB * cK * cD * 2);
  float*  mpf   = (float*)alloc((size_t)cB * cK * cD * 4);
  float*  cpf   = (float*)alloc((size_t)cB * cK * cNC * cD * 4);
  int* sidx   = (int*)alloc(cB * cK * 4);
  int* eidx   = (int*)alloc(cB * cK * 4);
  int* validf = (int*)alloc(cB * cK * 4);

  auto cvt = [&](const float* src, __bf16* dst, size_t n) {
    cvt_bf16_kernel<<<(unsigned)((n + 255) / 256), 256, 0, stream>>>(src, dst, (int)n);
  };
  auto gemm = [&](const __bf16* A, int lda, const __bf16* Bm, int ldb,
                  const float* bias, float alpha, int relu, float* Cf, int ldcf,
                  __bf16* Ch, int ldch, int M, int N, int Kd) {
    dim3 grid((M + GBM - 1) / GBM, (N + GBN - 1) / GBN);
    gemm_bf16_nt<<<grid, 256, 0, stream>>>(A, lda, Bm, ldb, bias, alpha, relu, Cf,
                                           ldcf, Ch, ldch, M, N, Kd);
  };

  // weight / candidate conversion to bf16
  cvt(attn_in_w, w_ain, (size_t)cNL * 3 * cD * cD);
  cvt(attn_out_w, w_aout, (size_t)cNL * cD * cD);
  cvt(ffn_w1, w_f1, (size_t)cNL * cDFF * cD);
  cvt(ffn_w2, w_f2, (size_t)cNL * cD * cDFF);
  cvt(span_w1, w_s1, (size_t)cD * 3 * cD);
  cvt(span_w2, w_s2, (size_t)cD * cD);
  cvt(mention_w, w_m, (size_t)cD * cD);
  cvt(entity_w, w_e, (size_t)cD * cD);
  cvt(cand, candb, (size_t)cB * cK * cNC * cD);

  // embeddings
  embed_kernel<<<(cBL * cD) / 256, 256, 0, stream>>>(token_ids, tok_emb, pos_emb, xf, xb);

  // encoder layers
  for (int l = 0; l < cNL; ++l) {
    gemm(xb, cD, w_ain + (size_t)l * 3 * cD * cD, cD, attn_in_b + l * 3 * cD, 1.f, 0,
         nullptr, 0, qkvb, 3 * cD, cBL, 3 * cD, cD);
    vt_kernel<<<(cB * cH * cHD * cL) / 256, 256, 0, stream>>>(qkvb, vtb);
    for (int b = 0; b < cB; ++b)
      for (int h = 0; h < cH; ++h) {
        const __bf16* Q = qkvb + (size_t)b * cL * 3 * cD + h * cHD;
        const __bf16* Kp = Q + cD;
        gemm(Q, 3 * cD, Kp, 3 * cD, nullptr, 0.125f, 0, sbuf, cL, nullptr, 0,
             cL, cL, cHD);
        softmax_kernel<<<cL, 256, 0, stream>>>(sbuf, pbuf, cL);
        const __bf16* VT = vtb + (size_t)(b * cH + h) * cHD * cL;
        __bf16* Odst = ob + (size_t)b * cL * cD + h * cHD;
        gemm(pbuf, cL, VT, cL, nullptr, 1.f, 0, nullptr, 0, Odst, cD, cL, cHD, cL);
      }
    gemm(ob, cD, w_aout + (size_t)l * cD * cD, cD, attn_out_b + l * cD, 1.f, 0,
         tmpf, cD, nullptr, 0, cBL, cD, cD);
    ln_kernel<<<cBL / 8, 256, 0, stream>>>(xf, tmpf, ln1_g + l * cD, ln1_b + l * cD, xf, xb);
    gemm(xb, cD, w_f1 + (size_t)l * cDFF * cD, cD, ffn_b1 + l * cDFF, 1.f, 1,
         nullptr, 0, h1b, cDFF, cBL, cDFF, cD);
    gemm(h1b, cDFF, w_f2 + (size_t)l * cD * cDFF, cDFF, ffn_b2 + l * cD, 1.f, 0,
         tmpf, cD, nullptr, 0, cBL, cD, cDFF);
    ln_kernel<<<cBL / 8, 256, 0, stream>>>(xf, tmpf, ln2_g + l * cD, ln2_b + l * cD, xf, xb);
  }

  // hidden output
  hipMemcpyAsync(out + HID_OFF, xf, (size_t)cBL * cD * sizeof(float),
                 hipMemcpyDeviceToDevice, stream);

  // heads
  logits_kernel<<<cBL / 8, 256, 0, stream>>>(xf, start_w, start_b, end_w, end_b,
                                             out + SL_OFF, out + EL_OFF);
  topk_kernel<<<1, 256, 0, stream>>>(out + SL_OFF, out + EL_OFF, sidx, eidx);
  feat_kernel<<<cB * cK, 256, 0, stream>>>(xf, sidx, eidx, featb, validf);
  gemm(featb, 3 * cD, w_s1, 3 * cD, span_b1, 1.f, 1, nullptr, 0, s1b, cD,
       cB * cK, cD, 3 * cD);
  gemm(s1b, cD, w_s2, cD, span_b2, 1.f, 0, spanraw, cD, nullptr, 0, cB * cK, cD, cD);
  compact_kernel<<<cB, 256, 0, stream>>>(spanraw, validf, out + SPAN_OFF, spanb);
  gemm(spanb, cD, w_m, cD, mention_b, 1.f, 0, mpf, cD, nullptr, 0, cB * cK, cD, cD);
  gemm(candb, cD, w_e, cD, entity_b, 1.f, 0, cpf, cD, nullptr, 0, cB * cK * cNC, cD, cD);
  scores_kernel<<<(cB * cK * cNC * 32) / 256, 256, 0, stream>>>(mpf, cpf, out + SC_OFF);
}
